// AttentionBlock_85143431676024
// MI455X (gfx1250) — matrix operand order repaired and checked by reading, not compiled, not run on hardware
//
#include <hip/hip_runtime.h>
#include <hip/hip_bf16.h>

// ---------------------------------------------------------------------------
// Types for gfx1250 WMMA (wave32): v_wmma_f32_16x16x32_bf16
//   A: 16x32 bf16  (16 bf16 / lane = 8 VGPRs)
//   B: 32x16 bf16  (16 bf16 / lane = 8 VGPRs)
//   C/D: 16x16 f32 (8 f32 / lane   = 8 VGPRs)
// ---------------------------------------------------------------------------
typedef __attribute__((ext_vector_type(8)))  float          v8f;
typedef __attribute__((ext_vector_type(16))) __bf16         v16bf;
typedef __attribute__((ext_vector_type(8)))  unsigned short u16x8;

union BFrag { v16bf bf; u16x8 h[2]; };

__device__ __forceinline__ unsigned short f2bf(float f) {
  unsigned u = __float_as_uint(f);
  u += 0x7fffu + ((u >> 16) & 1u);          // round-to-nearest-even
  return (unsigned short)(u >> 16);
}

__device__ __forceinline__ v8f vzero8() {
  v8f z;
#pragma unroll
  for (int i = 0; i < 8; ++i) z[i] = 0.f;
  return z;
}

__device__ __forceinline__ v8f wmma_bf16(const BFrag& a, const BFrag& b, v8f c) {
  // (neg_a, A, neg_b, B, c_mod, C, reuse_a, reuse_b)
  return __builtin_amdgcn_wmma_f32_16x16x32_bf16(false, a.bf, false, b.bf,
                                                 (short)0, c, false, false);
}

// Problem constants
#define BATCH 16
#define CCH   256
#define SEQ   1024
#define NH    4
#define DK    64
#define NPROJ 768   // 3*NH*DK

// ---------------------------------------------------------------------------
// Prep kernels: f32 -> bf16 with layout changes (tiny, memory-trivial).
// xs[b,s,c] = x[b,c,s];  wpT[n,k] = w_proj[k,n];  woT[n,k] = w_out[k,n]
// ---------------------------------------------------------------------------
__global__ __launch_bounds__(256) void k_prep_xs(const float* __restrict__ x,
                                                 unsigned short* __restrict__ xs) {
  int tid = blockIdx.x * 256 + threadIdx.x;     // BATCH*SEQ*CCH threads
  int c = tid & 255;
  int s = (tid >> 8) & 1023;
  int b = tid >> 18;
  xs[tid] = f2bf(x[((size_t)b * CCH + c) * SEQ + s]);
}

__global__ __launch_bounds__(256) void k_prep_wp(const float* __restrict__ wp,
                                                 unsigned short* __restrict__ wpT) {
  int tid = blockIdx.x * 256 + threadIdx.x;     // NPROJ*CCH threads
  int k = tid & 255;
  int n = tid >> 8;
  wpT[tid] = f2bf(wp[(size_t)k * NPROJ + n]);
}

__global__ __launch_bounds__(256) void k_prep_wo(const float* __restrict__ wo,
                                                 unsigned short* __restrict__ woT) {
  int tid = blockIdx.x * 256 + threadIdx.x;     // CCH*CCH threads
  int k = tid & 255;
  int n = tid >> 8;
  woT[tid] = f2bf(wo[(size_t)k * CCH + n]);
}

// ---------------------------------------------------------------------------
// QKV projection: per batch  [1024x256] @ [256x768] + bias.
// 8 waves/WG, each wave -> 16(M) x 64(N) tile, K loop of 8 x (k=32) WMMA.
// Writes Q[b,h,s,d], K[b,h,s,d] (d contiguous) and V transposed Vt[b,h,d,s]
// so the attention kernel's B-fragments are contiguous loads.
// ---------------------------------------------------------------------------
__global__ __launch_bounds__(256) void k_qkv(const unsigned short* __restrict__ xs,
                                             const unsigned short* __restrict__ wpT,
                                             const float* __restrict__ bproj,
                                             unsigned short* __restrict__ Qb,
                                             unsigned short* __restrict__ Kb,
                                             unsigned short* __restrict__ Vt) {
  const int lane = threadIdx.x & 31;
  const int wave = threadIdx.x >> 5;
  const int lr = lane & 15, lh = lane >> 4;
  const int b  = blockIdx.z;
  const int m0 = (blockIdx.x * 8 + wave) * 16;
  const int n0 = blockIdx.y * 64;

  v8f acc[4];
#pragma unroll
  for (int t = 0; t < 4; ++t) acc[t] = vzero8();

  const unsigned short* aRow = xs + (size_t)(b * SEQ + m0 + lr) * CCH;
#pragma unroll
  for (int k0 = 0; k0 < CCH; k0 += 32) {
    BFrag a;
    a.h[0] = *(const u16x8*)(aRow + k0 + lh * 8);
    a.h[1] = *(const u16x8*)(aRow + k0 + 16 + lh * 8);
#pragma unroll
    for (int t = 0; t < 4; ++t) {
      const unsigned short* bRow =
          wpT + (size_t)(n0 + t * 16 + lr) * CCH + k0 + lh * 8;
      BFrag bf;
      bf.h[0] = *(const u16x8*)(bRow);
      bf.h[1] = *(const u16x8*)(bRow + 16);
      acc[t] = wmma_bf16(a, bf, acc[t]);
    }
  }

#pragma unroll
  for (int t = 0; t < 4; ++t) {
    const int n  = n0 + t * 16 + lr;        // column in [0,768)
    const int hh = n / 192;                 // head (uniform per 16-col tile)
    const int r3 = n - hh * 192;            // q:[0,64) k:[64,128) v:[128,192)
    const float bias = bproj[n];
#pragma unroll
    for (int r = 0; r < 8; ++r) {
      const int row = m0 + r + lh * 8;      // C/D layout: M = r + 8*(lane/16)
      const unsigned short v = f2bf(acc[t][r] + bias);
      if (r3 < 64)
        Qb[((size_t)(b * NH + hh) * SEQ + row) * DK + r3] = v;
      else if (r3 < 128)
        Kb[((size_t)(b * NH + hh) * SEQ + row) * DK + (r3 - 64)] = v;
      else
        Vt[((size_t)(b * NH + hh) * DK + (r3 - 128)) * SEQ + row] = v;
    }
  }
}

// ---------------------------------------------------------------------------
// Flash attention per (batch, head): one wave owns a 16-row query tile,
// streams key/value j-tiles of 64 with online softmax (exp2 form).
// P is re-striped C-layout -> A-layout through a wave-private padded LDS tile.
// ---------------------------------------------------------------------------
__global__ __launch_bounds__(256) void k_attn(const unsigned short* __restrict__ Qb,
                                              const unsigned short* __restrict__ Kb,
                                              const unsigned short* __restrict__ Vt,
                                              unsigned short* __restrict__ obuf) {
  __shared__ __align__(16) unsigned short plds[8][16][72];  // 72 = 64 + pad (bank-conflict-free)

  const int lane = threadIdx.x & 31;
  const int wave = threadIdx.x >> 5;
  const int lr = lane & 15, lh = lane >> 4;
  const int b = blockIdx.z, hh = blockIdx.y;
  const int bh = b * NH + hh;
  const int i0 = blockIdx.x * 128 + wave * 16;
  const float SC = 0.125f * 1.44269504088896340736f;  // d_k^-0.5 * log2(e)

  // Resident Q A-fragments (16 rows x 64 d -> two k=32 fragments)
  BFrag qa[2];
  const unsigned short* qRow = Qb + ((size_t)bh * SEQ + i0 + lr) * DK;
#pragma unroll
  for (int kk = 0; kk < 2; ++kk) {
    qa[kk].h[0] = *(const u16x8*)(qRow + kk * 32 + lh * 8);
    qa[kk].h[1] = *(const u16x8*)(qRow + kk * 32 + 16 + lh * 8);
  }

  v8f o[4];
  float m[8], l[8];
#pragma unroll
  for (int dt = 0; dt < 4; ++dt) o[dt] = vzero8();
#pragma unroll
  for (int r = 0; r < 8; ++r) { m[r] = -1e30f; l[r] = 0.f; }

  for (int j0 = 0; j0 < SEQ; j0 += 64) {
    // ---- scores S = Q K^T (16 x 64) ----
    v8f s[4];
#pragma unroll
    for (int jt = 0; jt < 4; ++jt) {
      s[jt] = vzero8();
      const unsigned short* kRow = Kb + ((size_t)bh * SEQ + j0 + jt * 16 + lr) * DK;
#pragma unroll
      for (int kk = 0; kk < 2; ++kk) {
        BFrag kb;
        kb.h[0] = *(const u16x8*)(kRow + kk * 32 + lh * 8);
        kb.h[1] = *(const u16x8*)(kRow + kk * 32 + 16 + lh * 8);
        s[jt] = wmma_bf16(qa[kk], kb, s[jt]);
      }
    }
#pragma unroll
    for (int jt = 0; jt < 4; ++jt)
#pragma unroll
      for (int r = 0; r < 8; ++r) s[jt][r] *= SC;

    // ---- online softmax: row max over 64 cols (16-lane halves hold distinct rows) ----
    float rmax[8];
#pragma unroll
    for (int r = 0; r < 8; ++r)
      rmax[r] = fmaxf(fmaxf(s[0][r], s[1][r]), fmaxf(s[2][r], s[3][r]));
#pragma unroll
    for (int off = 1; off < 16; off <<= 1)
#pragma unroll
      for (int r = 0; r < 8; ++r)
        rmax[r] = fmaxf(rmax[r], __shfl_xor(rmax[r], off, 32));

    float corr[8], rsum[8];
#pragma unroll
    for (int r = 0; r < 8; ++r) {
      float mn = fmaxf(m[r], rmax[r]);
      corr[r] = exp2f(m[r] - mn);
      m[r] = mn;
      rsum[r] = 0.f;
    }
#pragma unroll
    for (int jt = 0; jt < 4; ++jt)
#pragma unroll
      for (int r = 0; r < 8; ++r) {
        float p = exp2f(s[jt][r] - m[r]);
        s[jt][r] = p;
        rsum[r] += p;
      }
#pragma unroll
    for (int off = 1; off < 16; off <<= 1)
#pragma unroll
      for (int r = 0; r < 8; ++r)
        rsum[r] += __shfl_xor(rsum[r], off, 32);
#pragma unroll
    for (int r = 0; r < 8; ++r) l[r] = l[r] * corr[r] + rsum[r];
#pragma unroll
    for (int dt = 0; dt < 4; ++dt)
#pragma unroll
      for (int r = 0; r < 8; ++r) o[dt][r] *= corr[r];

    // ---- P: C-layout -> A-layout via LDS (wave-private region) ----
    __syncthreads();
#pragma unroll
    for (int jt = 0; jt < 4; ++jt)
#pragma unroll
      for (int r = 0; r < 8; ++r)
        plds[wave][r + lh * 8][jt * 16 + lr] = f2bf(s[jt][r]);
    __syncthreads();

    BFrag pa[2];
#pragma unroll
    for (int kk = 0; kk < 2; ++kk) {
      pa[kk].h[0] = *(const u16x8*)&plds[wave][lr][kk * 32 + lh * 8];
      pa[kk].h[1] = *(const u16x8*)&plds[wave][lr][kk * 32 + 16 + lh * 8];
    }

    // ---- O += P @ V  (V stored transposed: contiguous B-fragment loads) ----
#pragma unroll
    for (int dt = 0; dt < 4; ++dt) {
      const unsigned short* vRow =
          Vt + ((size_t)bh * DK + dt * 16 + lr) * SEQ + j0;
#pragma unroll
      for (int kk = 0; kk < 2; ++kk) {
        BFrag vb;
        vb.h[0] = *(const u16x8*)(vRow + kk * 32 + lh * 8);
        vb.h[1] = *(const u16x8*)(vRow + kk * 32 + 16 + lh * 8);
        o[dt] = wmma_bf16(pa[kk], vb, o[dt]);
      }
    }
  }

  // ---- normalize and emit o[b,s, h*64+d] as bf16 ----
  float inv[8];
#pragma unroll
  for (int r = 0; r < 8; ++r) inv[r] = 1.f / l[r];
#pragma unroll
  for (int dt = 0; dt < 4; ++dt)
#pragma unroll
    for (int r = 0; r < 8; ++r) {
      const int row = i0 + r + lh * 8;
      const int col = hh * DK + dt * 16 + lr;
      obuf[(size_t)(b * SEQ + row) * CCH + col] = f2bf(o[dt][r] * inv[r]);
    }
}

// ---------------------------------------------------------------------------
// Output projection + bias + residual, writing transposed to [B,C,H,W].
// ---------------------------------------------------------------------------
__global__ __launch_bounds__(256) void k_outproj(const unsigned short* __restrict__ obuf,
                                                 const unsigned short* __restrict__ woT,
                                                 const float* __restrict__ bout,
                                                 const float* __restrict__ x,
                                                 float* __restrict__ out) {
  const int lane = threadIdx.x & 31;
  const int wave = threadIdx.x >> 5;
  const int lr = lane & 15, lh = lane >> 4;
  const int b  = blockIdx.z;
  const int m0 = (blockIdx.x * 8 + wave) * 16;
  const int n0 = blockIdx.y * 64;

  v8f acc[4];
#pragma unroll
  for (int t = 0; t < 4; ++t) acc[t] = vzero8();

  const unsigned short* aRow = obuf + (size_t)(b * SEQ + m0 + lr) * CCH;
#pragma unroll
  for (int k0 = 0; k0 < CCH; k0 += 32) {
    BFrag a;
    a.h[0] = *(const u16x8*)(aRow + k0 + lh * 8);
    a.h[1] = *(const u16x8*)(aRow + k0 + 16 + lh * 8);
#pragma unroll
    for (int t = 0; t < 4; ++t) {
      const unsigned short* bRow =
          woT + (size_t)(n0 + t * 16 + lr) * CCH + k0 + lh * 8;
      BFrag bf;
      bf.h[0] = *(const u16x8*)(bRow);
      bf.h[1] = *(const u16x8*)(bRow + 16);
      acc[t] = wmma_bf16(a, bf, acc[t]);
    }
  }

#pragma unroll
  for (int t = 0; t < 4; ++t) {
    const int c = n0 + t * 16 + lr;
    const float bias = bout[c];
#pragma unroll
    for (int r = 0; r < 8; ++r) {
      const int s = m0 + r + lh * 8;
      const size_t idx = ((size_t)b * CCH + c) * SEQ + s;
      out[idx] = acc[t][r] + bias + x[idx];   // residual = xs[b,s,c] = x[b,c,s]
    }
  }
}

// ---------------------------------------------------------------------------
extern "C" void kernel_launch(void* const* d_in, const int* in_sizes, int n_in,
                              void* d_out, int out_size, void* d_ws, size_t ws_size,
                              hipStream_t stream) {
  (void)in_sizes; (void)n_in; (void)out_size; (void)ws_size;
  const float* x     = (const float*)d_in[0];
  const float* wproj = (const float*)d_in[1];
  const float* bproj = (const float*)d_in[2];
  const float* wout  = (const float*)d_in[3];
  const float* bout  = (const float*)d_in[4];
  float* out = (float*)d_out;

  // Workspace layout (bf16 halves), total ~40.5 MB (fits easily in L2):
  unsigned short* ws  = (unsigned short*)d_ws;
  unsigned short* xs  = ws;                                       // [16,1024,256]
  unsigned short* wpT = xs  + (size_t)BATCH * SEQ * CCH;          // [768,256]
  unsigned short* woT = wpT + (size_t)NPROJ * CCH;                // [256,256]
  unsigned short* Qb  = woT + (size_t)CCH * CCH;                  // [16,4,1024,64]
  unsigned short* Kb  = Qb  + (size_t)BATCH * NH * SEQ * DK;      // [16,4,1024,64]
  unsigned short* Vt  = Kb  + (size_t)BATCH * NH * SEQ * DK;      // [16,4,64,1024]
  unsigned short* ob  = Vt  + (size_t)BATCH * NH * SEQ * DK;      // [16,1024,256]

  k_prep_xs<<<(BATCH * SEQ * CCH) / 256, 256, 0, stream>>>(x, xs);
  k_prep_wp<<<(NPROJ * CCH) / 256, 256, 0, stream>>>(wproj, wpT);
  k_prep_wo<<<(CCH * CCH) / 256, 256, 0, stream>>>(wout, woT);

  k_qkv    <<<dim3(SEQ / 128, NPROJ / 64, BATCH), 256, 0, stream>>>(xs, wpT, bproj, Qb, Kb, Vt);
  k_attn   <<<dim3(SEQ / 128, NH, BATCH),         256, 0, stream>>>(Qb, Kb, Vt, ob);
  k_outproj<<<dim3(SEQ / 128, CCH / 64, BATCH),   256, 0, stream>>>(ob, woT, bout, x, out);
}
